// RetBlock_9182640078867
// MI455X (gfx1250) — matrix operand order repaired and checked by reading, not compiled, not run on hardware
//
#include <hip/hip_runtime.h>
#include <hip/hip_bf16.h>
#include <stdint.h>

#define GEMM_USE_TDM 1

// ---------------------------------------------------------------------------
// Types for CDNA5 WMMA (wave32, 16x16x32 bf16 -> f32)
// ---------------------------------------------------------------------------
typedef __attribute__((ext_vector_type(16))) __bf16        v16bf;
typedef __attribute__((ext_vector_type(8)))  float         v8f;
typedef __attribute__((ext_vector_type(8)))  unsigned short ushort8;
typedef __attribute__((ext_vector_type(4)))  unsigned int  uint4v;
typedef __attribute__((ext_vector_type(8)))  unsigned int  uint8v;

union BfFrag { v16bf v; ushort8 h[2]; };

__device__ __forceinline__ v8f wmma_bf16(const BfFrag& a, const BfFrag& b, v8f c) {
  return __builtin_amdgcn_wmma_f32_16x16x32_bf16(false, a.v, false, b.v,
                                                 (short)0, c, false, false);
}

__device__ __forceinline__ unsigned short f2bf(float f) {
  unsigned u = __float_as_uint(f);
  unsigned r = u + 0x7FFFu + ((u >> 16) & 1u);   // round-to-nearest-even
  return (unsigned short)(r >> 16);
}

__device__ __forceinline__ float gelu_exact(float x) {
  return 0.5f * x * (1.0f + erff(x * 0.70710678118654752f));
}

#if GEMM_USE_TDM
// ---------------------------------------------------------------------------
// Tensor Data Mover: async 2D tile (rows x 32 bf16) Global -> LDS.
// D# packing per CDNA5 ISA ch.8: group0 = {count|flags, lds_addr, ga_lo,
// ga_hi|type=2}; group1 = {mask|data_size, dims/tile dims, strides}.
// Issued by one wave; completion via TENSORcnt.
// ---------------------------------------------------------------------------
__device__ __forceinline__ void tdm_load_tile_bf16(unsigned lds_addr,
                                                   const unsigned short* gsrc,
                                                   unsigned rows,
                                                   unsigned stride_elems) {
  unsigned long long ga = (unsigned long long)(uintptr_t)gsrc;
  uint4v g0;
  g0[0] = 1u;                                   // count=1, user mode
  g0[1] = lds_addr;                             // LDS byte address
  g0[2] = (unsigned)ga;                         // global_addr[31:0]
  g0[3] = (unsigned)(ga >> 32) | (2u << 30);    // global_addr[56:32] | type=2
  uint8v g1;
  g1[0] = 1u << 16;                             // workgroup_mask=0, data_size=2B
  g1[1] = 32u << 16;                            // tensor_dim0 = 32 (lo16 @ bit48)
  g1[2] = (rows & 0xFFFFu) << 16;               // tensor_dim0 hi | tensor_dim1 lo
  g1[3] = 32u << 16;                            // tensor_dim1 hi | tile_dim0 = 32
  g1[4] = rows & 0xFFFFu;                       // tile_dim1 = rows, tile_dim2 = 0
  g1[5] = stride_elems;                         // tensor_dim0_stride lo32
  g1[6] = 0u;                                   // stride hi16 | dim1_stride lo16
  g1[7] = 0u;                                   // dim1_stride hi32
  asm volatile("tensor_load_to_lds %0, %1" :: "s"(g0), "s"(g1) : "memory");
}
#endif

// ---------------------------------------------------------------------------
// K0: fp32 [K,N] -> bf16 transposed [N,K]
// ---------------------------------------------------------------------------
__global__ void cvt_transpose_bf16(const float* __restrict__ src,
                                   unsigned short* __restrict__ dst,
                                   int K, int N) {
  int idx = blockIdx.x * 256 + threadIdx.x;
  if (idx >= K * N) return;
  int k = idx / N, n = idx % N;
  dst[(size_t)n * K + k] = f2bf(src[idx]);
}

// ---------------------------------------------------------------------------
// K1: depthwise conv (ks=3 pad=1 or ks=5 pad=2), optional +input residual,
//     optional f32 and/or bf16 outputs.
// ---------------------------------------------------------------------------
__global__ void dwconv_k(const float* __restrict__ in, const float* __restrict__ w,
                         const float* __restrict__ bias,
                         float* __restrict__ outF, unsigned short* __restrict__ outBF,
                         int H, int W, int Cn, int ks, int pad, int add_input) {
  int pix = blockIdx.x;                 // b*H*W + h*W + w
  int hw  = pix % (H * W);
  int h   = hw / W, x = hw % W;
  int b   = pix / (H * W);
  for (int c = threadIdx.x; c < Cn; c += blockDim.x) {
    float acc = bias[c];
    for (int dy = 0; dy < ks; ++dy) {
      int hy = h + dy - pad;
      if (hy < 0 || hy >= H) continue;
      for (int dx = 0; dx < ks; ++dx) {
        int wx = x + dx - pad;
        if (wx < 0 || wx >= W) continue;
        acc += in[(((size_t)b * H + hy) * W + wx) * Cn + c] *
               w[(size_t)(dy * ks + dx) * Cn + c];
      }
    }
    size_t o = (size_t)pix * Cn + c;
    if (add_input) acc += in[o];
    if (outF)  outF[o]  = acc;
    if (outBF) outBF[o] = f2bf(acc);
  }
}

// ---------------------------------------------------------------------------
// K2: LayerNorm over C=256 -> bf16 output. One block (256 thr / 8 waves) per row.
// ---------------------------------------------------------------------------
__global__ __launch_bounds__(256) void ln_bf16(const float* __restrict__ in,
                                               const float* __restrict__ g,
                                               const float* __restrict__ bta,
                                               unsigned short* __restrict__ out,
                                               int C) {
  __shared__ float red[16];
  int row = blockIdx.x, tid = threadIdx.x;
  float x = in[(size_t)row * C + tid];
  float s = x, q = x * x;
  for (int d = 16; d >= 1; d >>= 1) {
    s += __shfl_xor(s, d, 32);
    q += __shfl_xor(q, d, 32);
  }
  int wid = tid >> 5;
  if ((tid & 31) == 0) { red[wid] = s; red[wid + 8] = q; }
  __syncthreads();
  float ts = 0.f, tq = 0.f;
  for (int i = 0; i < 8; ++i) { ts += red[i]; tq += red[i + 8]; }
  float mean = ts / C;
  float var  = tq / C - mean * mean;
  float r    = rsqrtf(var + 1e-6f);
  out[(size_t)row * C + tid] = f2bf((x - mean) * r * g[tid] + bta[tid]);
}

// ---------------------------------------------------------------------------
// K3: WMMA GEMM  out[M,N] = act((A[M,K] @ Bt[N,K]^T + bias) * scale) + res
//     A, Bt in bf16(ushort bits). Block tile 128x64, 8 waves of 32x32.
//     Staging: TDM tensor_load_to_lds, double-buffered, 1 barrier / K-step.
// ---------------------------------------------------------------------------
__global__ __launch_bounds__(256) void gemm_bf16_wmma(
    const unsigned short* __restrict__ A, const unsigned short* __restrict__ Bt,
    const float* __restrict__ bias, const float* __restrict__ res,
    float* __restrict__ out, int M, int Nn, int K, float scale, int act) {
  __shared__ __align__(16) unsigned short As[2][128][32];
  __shared__ __align__(16) unsigned short Bs[2][64][32];
  int tid = threadIdx.x;
  int lane = tid & 31, wid = tid >> 5;
  int wr = wid & 3, wc = wid >> 2;              // 4x2 wave grid
  int hi = lane >> 4, lr = lane & 15;
  int bm = blockIdx.x * 128, bn = blockIdx.y * 64;
  const int kbA = hi ? 16 : 0;                  // A-operand K base per half-wave
  const int kbB = hi ? 16 : 0;                  // B-operand K base per half-wave

  v8f acc[2][2];
#pragma unroll
  for (int i = 0; i < 2; ++i)
#pragma unroll
    for (int j = 0; j < 2; ++j) acc[i][j] = {};

  const int nk = K >> 5;

#if GEMM_USE_TDM
  unsigned ldsA[2] = { (unsigned)(size_t)&As[0][0][0], (unsigned)(size_t)&As[1][0][0] };
  unsigned ldsB[2] = { (unsigned)(size_t)&Bs[0][0][0], (unsigned)(size_t)&Bs[1][0][0] };
  const unsigned short* Abase = A  + (size_t)bm * K;
  const unsigned short* Bbase = Bt + (size_t)bn * K;
  if (wid == 0) {  // one wave drives the DMA engine (EXEC-independent op)
    tdm_load_tile_bf16(ldsA[0], Abase, 128, (unsigned)K);
    tdm_load_tile_bf16(ldsB[0], Bbase, 64,  (unsigned)K);
  }
  for (int it = 0; it < nk; ++it) {
    int cur = it & 1;
    if (wid == 0) __builtin_amdgcn_s_wait_tensorcnt(0);  // tile `it` landed
    __syncthreads();                                     // visible to all waves
    if (wid == 0 && it + 1 < nk) {                       // prefetch next tile
      size_t ko = (size_t)(it + 1) << 5;
      tdm_load_tile_bf16(ldsA[cur ^ 1], Abase + ko, 128, (unsigned)K);
      tdm_load_tile_bf16(ldsB[cur ^ 1], Bbase + ko, 64,  (unsigned)K);
    }
    BfFrag af[2], bfr[2];
#pragma unroll
    for (int i = 0; i < 2; ++i) {
      int r = wr * 32 + i * 16 + lr;
      af[i].h[0] = *(const ushort8*)&As[cur][r][kbA];
      af[i].h[1] = *(const ushort8*)&As[cur][r][kbA + 8];
    }
#pragma unroll
    for (int j = 0; j < 2; ++j) {
      int c = wc * 32 + j * 16 + lr;
      bfr[j].h[0] = *(const ushort8*)&Bs[cur][c][kbB];
      bfr[j].h[1] = *(const ushort8*)&Bs[cur][c][kbB + 8];
    }
#pragma unroll
    for (int i = 0; i < 2; ++i)
#pragma unroll
      for (int j = 0; j < 2; ++j) acc[i][j] = wmma_bf16(af[i], bfr[j], acc[i][j]);
    // no trailing barrier: next iteration's barrier orders these completed LDS
    // reads before the DMA that overwrites this buffer (issued in the next step, after that barrier).
  }
#else
  for (int it = 0; it < nk; ++it) {
    int kt = it << 5;
#pragma unroll
    for (int i = 0; i < 2; ++i) {
      int idx = tid + i * 256;
      int row = idx >> 2, seg = (idx & 3) * 8;
      *(ushort8*)&As[0][row][seg] =
          *(const ushort8*)&A[(size_t)(bm + row) * K + kt + seg];
    }
    {
      int row = tid >> 2, seg = (tid & 3) * 8;
      *(ushort8*)&Bs[0][row][seg] =
          *(const ushort8*)&Bt[(size_t)(bn + row) * K + kt + seg];
    }
    __syncthreads();
    BfFrag af[2], bfr[2];
#pragma unroll
    for (int i = 0; i < 2; ++i) {
      int r = wr * 32 + i * 16 + lr;
      af[i].h[0] = *(const ushort8*)&As[0][r][kbA];
      af[i].h[1] = *(const ushort8*)&As[0][r][kbA + 8];
    }
#pragma unroll
    for (int j = 0; j < 2; ++j) {
      int c = wc * 32 + j * 16 + lr;
      bfr[j].h[0] = *(const ushort8*)&Bs[0][c][kbB];
      bfr[j].h[1] = *(const ushort8*)&Bs[0][c][kbB + 8];
    }
#pragma unroll
    for (int i = 0; i < 2; ++i)
#pragma unroll
      for (int j = 0; j < 2; ++j) acc[i][j] = wmma_bf16(af[i], bfr[j], acc[i][j]);
    __syncthreads();
  }
#endif

  // epilogue: bias -> scale -> gelu -> residual
#pragma unroll
  for (int i = 0; i < 2; ++i)
#pragma unroll
    for (int j = 0; j < 2; ++j) {
      int n  = bn + wc * 32 + j * 16 + lr;
      float bv = bias ? bias[n] : 0.f;
#pragma unroll
      for (int r = 0; r < 8; ++r) {
        int m = bm + wr * 32 + i * 16 + r + hi * 8;
        float v = (acc[i][j][r] + bv) * scale;
        if (act) v = gelu_exact(v);
        size_t o = (size_t)m * Nn + n;
        if (res) v += res[o];
        out[o] = v;
      }
    }
}

// ---------------------------------------------------------------------------
// K4: theta-shift (RoPE-pair rotate) on q,k and pack q,k,v to bf16 layouts:
//     qr,kr: [B,N,L,D] row-major ; vT: [B,N,D,L] (K-contiguous for WMMA B-op)
// ---------------------------------------------------------------------------
__global__ void theta_pack(const float* __restrict__ q, const float* __restrict__ k,
                           const float* __restrict__ v,
                           const float* __restrict__ sn, const float* __restrict__ cs,
                           unsigned short* __restrict__ qr, unsigned short* __restrict__ kr,
                           unsigned short* __restrict__ vT,
                           int L, int C, int Nh, int D) {
  size_t idx = (size_t)blockIdx.x * 256 + threadIdx.x;
  int c = (int)(idx % C);
  size_t bl = idx / C;
  int l = (int)(bl % L);
  int b = (int)(bl / L);
  int n = c / D, d = c % D;
  float sv = sn[(size_t)l * D + d], cv = cs[(size_t)l * D + d];
  float qv = q[idx], kv = k[idx], vv = v[idx];
  int dp = (d & 1) ? (d - 1) : (d + 1);
  size_t pidx = idx - d + dp;
  float qrot = (d & 1) ? q[pidx] : -q[pidx];
  float krot = (d & 1) ? k[pidx] : -k[pidx];
  size_t ro = (((size_t)(b * Nh + n)) * L + l) * D + d;
  qr[ro] = f2bf(qv * cv + qrot * sv);
  kr[ro] = f2bf(kv * cv + krot * sv);
  vT[(((size_t)(b * Nh + n)) * D + d) * L + l] = f2bf(vv);
}

// ---------------------------------------------------------------------------
// K5: flash attention. One 16-row query tile per wave; key loop in steps of 32.
//     S = Q(16x32) @ K^T via one wmma per 16-key half; streaming softmax with
//     mask; P staged via per-wave LDS into A-operand layout; O += P @ V via
//     two wmmas (V pre-transposed [B,N,D,L]). Output to [B,L,C] fp32.
// ---------------------------------------------------------------------------
__global__ __launch_bounds__(256) void attn_flash(
    const unsigned short* __restrict__ qr, const unsigned short* __restrict__ kr,
    const unsigned short* __restrict__ vT, const float* __restrict__ mask,
    float* __restrict__ outp, int Nh, int L, int Ctot) {
  __shared__ __align__(16) unsigned short plds[8][16][32];
  const int D = 32;
  int tid = threadIdx.x, lane = tid & 31, w = tid >> 5;
  int hi = lane >> 4, lr = lane & 15;
  int tilesPerBN = L / 16;
  int gt = blockIdx.x * 8 + w;
  int qt = gt % tilesPerBN;
  int bn = gt / tilesPerBN;
  int n  = bn % Nh;
  int b  = bn / Nh;

  const unsigned short* qb = qr + (size_t)bn * L * D;
  const unsigned short* kb = kr + (size_t)bn * L * D;
  const unsigned short* vb = vT + (size_t)bn * D * L;
  const float* mrow = mask + (size_t)n * L * L;

  int qrow0 = qt * 16;
  BfFrag qf;
  {
    const unsigned short* p = qb + (size_t)(qrow0 + lr) * D + (hi ? 16 : 0);
    qf.h[0] = *(const ushort8*)p;
    qf.h[1] = *(const ushort8*)(p + 8);
  }

  float mstate[8], lstate[8];
#pragma unroll
  for (int r = 0; r < 8; ++r) { mstate[r] = -3.0e38f; lstate[r] = 0.f; }
  v8f o0 = {}, o1 = {};
  v8f zero = {};

  for (int kt0 = 0; kt0 < L; kt0 += 32) {
    // K^T B-operands: lane = key row, 16 contiguous D values
    BfFrag kf0, kf1;
    {
      const unsigned short* p = kb + (size_t)(kt0 + lr) * D + (hi ? 16 : 0);
      kf0.h[0] = *(const ushort8*)p;
      kf0.h[1] = *(const ushort8*)(p + 8);
      p += (size_t)16 * D;
      kf1.h[0] = *(const ushort8*)p;
      kf1.h[1] = *(const ushort8*)(p + 8);
    }
    v8f s0 = wmma_bf16(qf, kf0, zero);
    v8f s1 = wmma_bf16(qf, kf1, zero);

    // mask add + streaming softmax (per row r within this half-wave)
#pragma unroll
    for (int r = 0; r < 8; ++r) {
      int qrow = qrow0 + r + hi * 8;
      const float* mp = mrow + (size_t)qrow * L + kt0 + lr;
      float a0 = s0[r] + mp[0];
      float a1 = s1[r] + mp[16];
      float mt = fmaxf(a0, a1);
      for (int d = 8; d >= 1; d >>= 1) mt = fmaxf(mt, __shfl_xor(mt, d, 32));
      float mnew  = fmaxf(mstate[r], mt);
      float alpha = __expf(mstate[r] - mnew);
      float p0 = __expf(a0 - mnew), p1 = __expf(a1 - mnew);
      float ls = p0 + p1;
      for (int d = 8; d >= 1; d >>= 1) ls += __shfl_xor(ls, d, 32);
      lstate[r] = lstate[r] * alpha + ls;
      mstate[r] = mnew;
      o0[r] *= alpha;
      o1[r] *= alpha;
      int prow = r + hi * 8;
      plds[w][prow][lr]      = f2bf(p0);
      plds[w][prow][lr + 16] = f2bf(p1);
    }
    __builtin_amdgcn_wave_barrier();   // DS ops in-order per wave; keep schedule

    // P as A-operand (layout transpose through LDS)
    BfFrag pf;
    {
      int kbp = hi ? 16 : 0;
      pf.h[0] = *(const ushort8*)&plds[w][lr][kbp];
      pf.h[1] = *(const ushort8*)&plds[w][lr][kbp + 8];
    }
    // V B-operands from transposed layout: lane = D column, contiguous along L
    BfFrag vf0, vf1;
    {
      int koff = kt0 + (hi ? 16 : 0);
      const unsigned short* p0 = vb + (size_t)lr * L + koff;
      vf0.h[0] = *(const ushort8*)p0;
      vf0.h[1] = *(const ushort8*)(p0 + 8);
      const unsigned short* p1 = vb + (size_t)(lr + 16) * L + koff;
      vf1.h[0] = *(const ushort8*)p1;
      vf1.h[1] = *(const ushort8*)(p1 + 8);
    }
    o0 = wmma_bf16(pf, vf0, o0);
    o1 = wmma_bf16(pf, vf1, o1);
  }

  // normalize + write [B, L, C] with C = Nh*D
#pragma unroll
  for (int r = 0; r < 8; ++r) {
    int qrow = qrow0 + r + hi * 8;
    float inv = 1.f / lstate[r];
    size_t base = ((size_t)b * L + qrow) * Ctot + n * D;
    outp[base + lr]      = o0[r] * inv;
    outp[base + 16 + lr] = o1[r] * inv;
  }
}

// ---------------------------------------------------------------------------
// K6: out_bf16 = bf16(a + b)
// ---------------------------------------------------------------------------
__global__ void add_cvt_bf16(const float* __restrict__ a, const float* __restrict__ b,
                             unsigned short* __restrict__ out, size_t n) {
  size_t i = (size_t)blockIdx.x * 256 + threadIdx.x;
  if (i < n) out[i] = f2bf(a[i] + b[i]);
}

// ---------------------------------------------------------------------------
// Host orchestration
// ---------------------------------------------------------------------------
extern "C" void kernel_launch(void* const* d_in, const int* in_sizes, int n_in,
                              void* d_out, int out_size, void* d_ws, size_t ws_size,
                              hipStream_t stream) {
  (void)in_sizes; (void)n_in; (void)out_size; (void)ws_size;
  const int B = 2, H = 56, W = 56, C = 256, Nh = 8, D = 32, F = 1024;
  const int L = H * W;            // 3136
  const int M = B * L;            // 6272
  const float SCALING = 0.17677669529663689f; // 32^-0.5

  const float* x_in   = (const float*)d_in[0];
  const float* sn     = (const float*)d_in[1];
  const float* cs     = (const float*)d_in[2];
  const float* mask   = (const float*)d_in[3];
  const float* pos_w  = (const float*)d_in[4];
  const float* pos_b  = (const float*)d_in[5];
  const float* ln1_g  = (const float*)d_in[6];
  const float* ln1_b  = (const float*)d_in[7];
  const float* wq     = (const float*)d_in[8];
  const float* bq     = (const float*)d_in[9];
  const float* wk     = (const float*)d_in[10];
  const float* bk     = (const float*)d_in[11];
  const float* wv     = (const float*)d_in[12];
  const float* bv     = (const float*)d_in[13];
  const float* lepe_w = (const float*)d_in[14];
  const float* lepe_b = (const float*)d_in[15];
  const float* wo     = (const float*)d_in[16];
  const float* bo     = (const float*)d_in[17];
  const float* ln2_g  = (const float*)d_in[18];
  const float* ln2_b  = (const float*)d_in[19];
  const float* w1     = (const float*)d_in[20];
  const float* b1     = (const float*)d_in[21];
  const float* ffn_w  = (const float*)d_in[22];
  const float* ffn_b  = (const float*)d_in[23];
  const float* w2     = (const float*)d_in[24];
  const float* b2     = (const float*)d_in[25];
  float* out = (float*)d_out;

  // ---- workspace carve-up (256B aligned) ----
  uint8_t* ws = (uint8_t*)d_ws;
  size_t off = 0;
  auto alloc = [&](size_t bytes) -> uint8_t* {
    uint8_t* p = ws + off;
    off += (bytes + 255) & ~(size_t)255;
    return p;
  };
  unsigned short* wqT = (unsigned short*)alloc((size_t)C * C * 2);
  unsigned short* wkT = (unsigned short*)alloc((size_t)C * C * 2);
  unsigned short* wvT = (unsigned short*)alloc((size_t)C * C * 2);
  unsigned short* woT = (unsigned short*)alloc((size_t)C * C * 2);
  unsigned short* w1T = (unsigned short*)alloc((size_t)C * F * 2);
  unsigned short* w2T = (unsigned short*)alloc((size_t)F * C * 2);
  float*          xbuf = (float*)alloc((size_t)M * C * 4);
  unsigned short* ybuf = (unsigned short*)alloc((size_t)M * C * 2); // LN1/LN2 out
  float*          bigA = (float*)alloc((size_t)M * F * 4);          // q|k|v|attn, later z1
  float* qbuf = bigA;
  float* kbuf = bigA + (size_t)M * C;
  float* vbuf = bigA + (size_t)2 * M * C;
  float* attn = bigA + (size_t)3 * M * C;
  unsigned short* qrB = (unsigned short*)alloc((size_t)B * Nh * L * D * 2);
  unsigned short* krB = (unsigned short*)alloc((size_t)B * Nh * L * D * 2);
  unsigned short* vTB = (unsigned short*)alloc((size_t)B * Nh * D * L * 2);
  float*          lepe = (float*)alloc((size_t)M * C * 4);
  unsigned short* pbuf = (unsigned short*)alloc((size_t)M * C * 2);
  unsigned short* z2b  = (unsigned short*)alloc((size_t)M * F * 2);

  // ---- 0) weight conversion (bf16, transposed [N,K]) ----
  cvt_transpose_bf16<<<(C * C + 255) / 256, 256, 0, stream>>>(wq, wqT, C, C);
  cvt_transpose_bf16<<<(C * C + 255) / 256, 256, 0, stream>>>(wk, wkT, C, C);
  cvt_transpose_bf16<<<(C * C + 255) / 256, 256, 0, stream>>>(wv, wvT, C, C);
  cvt_transpose_bf16<<<(C * C + 255) / 256, 256, 0, stream>>>(wo, woT, C, C);
  cvt_transpose_bf16<<<(C * F + 255) / 256, 256, 0, stream>>>(w1, w1T, C, F);
  cvt_transpose_bf16<<<(F * C + 255) / 256, 256, 0, stream>>>(w2, w2T, F, C);

  // ---- 1) x = x + dwconv3x3(x) ----
  dwconv_k<<<B * H * W, 256, 0, stream>>>(x_in, pos_w, pos_b, xbuf, nullptr,
                                          H, W, C, 3, 1, 1);
  // ---- 2) y = LN1(x) -> bf16 ----
  ln_bf16<<<M, 256, 0, stream>>>(xbuf, ln1_g, ln1_b, ybuf, C);

  // ---- 3) q,k,v GEMMs (WMMA + TDM staging) ----
  dim3 gqkv(M / 128, C / 64);
  gemm_bf16_wmma<<<gqkv, 256, 0, stream>>>(ybuf, wqT, bq, nullptr, qbuf, M, C, C, 1.0f, 0);
  gemm_bf16_wmma<<<gqkv, 256, 0, stream>>>(ybuf, wkT, bk, nullptr, kbuf, M, C, C, SCALING, 0);
  gemm_bf16_wmma<<<gqkv, 256, 0, stream>>>(ybuf, wvT, bv, nullptr, vbuf, M, C, C, 1.0f, 0);

  // ---- 4) lepe = dwconv5x5(v) ----
  dwconv_k<<<B * H * W, 256, 0, stream>>>(vbuf, lepe_w, lepe_b, lepe, nullptr,
                                          H, W, C, 5, 2, 0);
  // ---- 5) theta shift + pack ----
  theta_pack<<<(int)(((size_t)M * C) / 256), 256, 0, stream>>>(
      qbuf, kbuf, vbuf, sn, cs, qrB, krB, vTB, L, C, Nh, D);

  // ---- 6) flash attention (WMMA) ----
  attn_flash<<<(B * Nh * (L / 16)) / 8, 256, 0, stream>>>(
      qrB, krB, vTB, mask, attn, Nh, L, C);

  // ---- 7) (attn + lepe) -> bf16 ; x = x + (.)@wo + bo ----
  add_cvt_bf16<<<(int)(((size_t)M * C + 255) / 256), 256, 0, stream>>>(
      attn, lepe, pbuf, (size_t)M * C);
  gemm_bf16_wmma<<<gqkv, 256, 0, stream>>>(pbuf, woT, bo, xbuf, xbuf, M, C, C, 1.0f, 0);

  // ---- 8) z = LN2(x); z1 = gelu(z@w1+b1) ----
  ln_bf16<<<M, 256, 0, stream>>>(xbuf, ln2_g, ln2_b, ybuf, C);
  float* z1 = bigA;  // q/k/v/attn are dead now; [M,F] fits exactly
  dim3 gffn1(M / 128, F / 64);
  gemm_bf16_wmma<<<gffn1, 256, 0, stream>>>(ybuf, w1T, b1, nullptr, z1, M, F, C, 1.0f, 1);

  // ---- 9) z2 = z1 + dwconv3x3(z1) -> bf16 ----
  dwconv_k<<<B * H * W, 256, 0, stream>>>(z1, ffn_w, ffn_b, nullptr, z2b,
                                          H, W, F, 3, 1, 1);
  // ---- 10) out = x + z2@w2 + b2 ----
  dim3 gffn2(M / 128, C / 64);
  gemm_bf16_wmma<<<gffn2, 256, 0, stream>>>(z2b, w2T, b2, xbuf, out, M, C, F, 1.0f, 0);
}
